// ODERGRU_31447750541940
// MI455X (gfx1250) — matrix operand order repaired and checked by reading, not compiled, not run on hardware
//
#include <hip/hip_runtime.h>
#include <math.h>

// ---------------- types ----------------
typedef __bf16 bhalf;
typedef __attribute__((ext_vector_type(16))) __bf16 v16bf;
typedef __attribute__((ext_vector_type(8)))  float  v8f;

// ---------------- model dims ----------------
#define BB    64
#define SS    50
#define CCH   3
#define NF    129
#define TT    29
#define MFB   32
#define DD    16
#define MOFF  120
#define HID   136
#define ODEU  256
#define NCLS  5
#define CIN   96      // CCH*MFB
#define T1    25      // TT - 4
#define T2    21      // T1 - 4
#define NSEQ  3200    // BB*SS
#define KCONV 480     // CIN*5

// ---------------- workspace layout (bytes) ----------------
#define OFF_FB    0u           // bf16 [32][160]
#define OFF_C1T   16384u       // bf16 [96][480]
#define OFF_C2T   110592u      // bf16 [16][480]
#define OFF_O0    131072u      // bf16 [256][160]
#define OFF_O1    212992u      // bf16 [256][256]
#define OFF_O2    344064u      // bf16 [256][256]
#define OFF_O3    475136u      // bf16 [136][256]
#define OFF_ST1   544768u      // f32  [2*96]
#define OFF_ST2   545536u      // f32  [2*16]
#define OFF_H0    1048576u     // bf16 [3200][96][32]   (also reused as h1)
#define OFF_BIG   20709376u    // f32  conv1raw [3200][96][25] (30.72MB)
#define OFF_C2RAW OFF_BIG                       // f32 [3200][16][21] (aliases dead conv1raw)
#define OFF_FEATS (OFF_BIG + 4300800u)          // f32 [3200][21][16]
#define OFF_XDXO  (OFF_BIG + 8601600u)          // f32 [3200][136]

#define SCAN_LDS_BYTES 141312   // (1024+7680+10240)*4 + 2*64*256*2

// ---------------- bf16 helpers ----------------
static __device__ __forceinline__ bhalf f2bf(float f) {
  unsigned u = __builtin_bit_cast(unsigned, f);
  u += 0x7FFFu + ((u >> 16) & 1u);                 // round-to-nearest-even
  unsigned short h = (unsigned short)(u >> 16);
  return __builtin_bit_cast(bhalf, h);
}

static __device__ __forceinline__ float sigm(float x) { return 1.0f / (1.0f + expf(-x)); }

// A-matrix 16x32 bf16 element pattern (ISA 7.12.2):
// lanes 0-15: rows M, K = {0..7, 16..23}; lanes 16-31: same rows, K = {8..15, 24..31}
static __device__ __forceinline__ int kpat_rel(int e, int half) {
  int v = e >> 1, p = e & 1;
  return (v < 4) ? (8 * half + 2 * v + p) : (16 + 8 * half + 2 * (v - 4) + p);
}

static __device__ __forceinline__ v8f wmma_bf(v16bf a, v16bf b, v8f c) {
  return __builtin_amdgcn_wmma_f32_16x16x32_bf16(false, a, false, b, (short)0, c, false, false);
}

// A fragment from an LDS/global bf16 row-major tile (row stride lda elems), k-chunk at k0
static __device__ __forceinline__ v16bf fragA_mem(const bhalf* A, int lda, int mbase, int k0) {
  int lane = threadIdx.x & 31;
  int half = lane >> 4, r = lane & 15;
  const bhalf* p = A + (long)(mbase + r) * lda + k0;
  v16bf a;
#pragma unroll
  for (int e = 0; e < 16; ++e) a[e] = p[kpat_rel(e, half)];
  return a;
}

// B fragment: weights stored [N][K] (k contiguous).  lanes 0-15: col N, K {0..7, 16..23};
// lanes 16-31: col N, K {8..15, 24..31} (same k pattern as A).
// Branchless: clamp row, select-zero after the load (no exec-mask churn).
static __device__ __forceinline__ v16bf fragB_glb(const bhalf* B, int ldk, int n0, int k0, int nmax) {
  int lane = threadIdx.x & 31;
  int half = lane >> 4, c = lane & 15;
  int n = n0 + c;
  bool ok = n < nmax;
  int nc = ok ? n : (nmax - 1);
  const bhalf* p = B + (long)nc * ldk + k0;
  bhalf z = f2bf(0.0f);
  v16bf b;
#pragma unroll
  for (int e = 0; e < 16; ++e) {
    bhalf v = p[kpat_rel(e, half)];
    b[e] = ok ? v : z;
  }
  return b;
}

// ---------------- K0: weight prep (f32 -> bf16, transpose/pad) ----------------
__global__ void k_prep(const float* fw, const float* s_tri, const float* c1w, const float* c2w,
                       const float* ow0, const float* ow1, const float* ow2, const float* ow3,
                       bhalf* wfb, bhalf* c1t, bhalf* c2t,
                       bhalf* o0t, bhalf* o1t, bhalf* o2t, bhalf* o3t) {
  int stride = gridDim.x * blockDim.x;
  int tid0 = blockIdx.x * blockDim.x + threadIdx.x;
  for (int i = tid0; i < 32 * 160; i += stride) {            // W = fw * s_tri, [m][f pad160]
    int m = i / 160, f = i % 160;
    wfb[i] = f2bf(f < NF ? fw[f * 32 + m] * s_tri[f * 32 + m] : 0.0f);
  }
  for (int i = tid0; i < 96 * KCONV; i += stride) c1t[i] = f2bf(c1w[i]);   // already [co][ci*5+dt]
  for (int i = tid0; i < 16 * KCONV; i += stride) c2t[i] = f2bf(c2w[i]);
  for (int i = tid0; i < 256 * 160; i += stride) {           // [u][h pad160]
    int u = i / 160, h = i % 160;
    o0t[i] = f2bf(h < HID ? ow0[h * 256 + u] : 0.0f);
  }
  for (int i = tid0; i < 256 * 256; i += stride) {           // [u2][u1]
    int u2 = i / 256, u1 = i % 256;
    o1t[i] = f2bf(ow1[u1 * 256 + u2]);
    o2t[i] = f2bf(ow2[u1 * 256 + u2]);
  }
  for (int i = tid0; i < 136 * 256; i += stride) {           // [h][u]
    int h = i / 256, u = i % 256;
    o3t[i] = f2bf(ow3[u * 136 + h]);
  }
}

// ---------------- K1: filterbank einsum as WMMA GEMM ----------------
// per (n,c): C[t][m] = sum_f x[n,c,f,t] * W[f][m];  M=32(t pad), N=32, K=160(pad)
// Branchless gather: clamp f/t so every load is valid, then cndmask to zero.
static __device__ __forceinline__ v16bf fragA_einsum(const float* x, int n, int c, int t0, int k0) {
  int lane = threadIdx.x & 31;
  int half = lane >> 4;
  int t = t0 + (lane & 15);
  bool tok = t < TT;
  int tc = tok ? t : (TT - 1);
  long base = (long)(n * CCH + c) * NF * TT;
  v16bf a;
#pragma unroll
  for (int e = 0; e < 16; ++e) {
    int f = k0 + kpat_rel(e, half);
    bool fok = f < NF;
    int fc = fok ? f : (NF - 1);
    float v = x[base + (long)fc * TT + tc];
    v = (tok && fok) ? v : 0.0f;
    a[e] = f2bf(v);
  }
  return a;
}

__global__ void k_einsum(const float* __restrict__ x, const bhalf* __restrict__ wfb,
                         bhalf* __restrict__ h0) {
  int wid = (blockIdx.x * blockDim.x + threadIdx.x) >> 5;
  if (wid >= NSEQ * CCH) return;
  int n = wid / CCH, c = wid % CCH;
  v8f acc[2][2] = {};
  for (int kc = 0; kc < 5; ++kc) {
    int k0 = kc * 32;
    v16bf a0 = fragA_einsum(x, n, c, 0, k0);
    v16bf a1 = fragA_einsum(x, n, c, 16, k0);
    v16bf b0 = fragB_glb(wfb, 160, 0, k0, 32);
    v16bf b1 = fragB_glb(wfb, 160, 16, k0, 32);
    acc[0][0] = wmma_bf(a0, b0, acc[0][0]);
    acc[0][1] = wmma_bf(a0, b1, acc[0][1]);
    acc[1][0] = wmma_bf(a1, b0, acc[1][0]);
    acc[1][1] = wmma_bf(a1, b1, acc[1][1]);
  }
  int lane = threadIdx.x & 31, half = lane >> 4, cl = lane & 15;
#pragma unroll
  for (int rt = 0; rt < 2; ++rt)
#pragma unroll
    for (int ct = 0; ct < 2; ++ct)
#pragma unroll
      for (int rr = 0; rr < 8; ++rr) {
        int t = rt * 16 + rr + 8 * half;
        int m = ct * 16 + cl;
        if (t < TT) h0[((long)n * CIN + c * MFB + m) * 32 + t] = f2bf(acc[rt][ct][rr]);
      }
}

// ---------------- K2/K5: conv1d as im2col WMMA GEMM ----------------
// Branchless A gather: clamp t so loads are always in-bounds, cndmask to zero.
static __device__ __forceinline__ v16bf fragA_conv(const bhalf* hsrc, int n, int t0, int k0, int tmax) {
  int lane = threadIdx.x & 31;
  int half = lane >> 4;
  int t = t0 + (lane & 15);
  bool tok = t < tmax;
  int tc = tok ? t : (tmax - 1);
  const bhalf* base = hsrc + (long)n * CIN * 32;
  bhalf z = f2bf(0.0f);
  v16bf a;
#pragma unroll
  for (int e = 0; e < 16; ++e) {
    int k = k0 + kpat_rel(e, half);
    int ci = k / 5, dt = k % 5;
    bhalf v = base[ci * 32 + tc + dt];
    a[e] = tok ? v : z;
  }
  return a;
}

__global__ void k_conv1(const bhalf* __restrict__ h0, const bhalf* __restrict__ c1t,
                        const float* __restrict__ bias, float* __restrict__ out) {
  int wid = (blockIdx.x * blockDim.x + threadIdx.x) >> 5;
  if (wid >= NSEQ * 2) return;
  int n = wid >> 1, rt = wid & 1;
  v8f acc[6] = {};
  for (int kc = 0; kc < 15; ++kc) {
    int k0 = kc * 32;
    v16bf a = fragA_conv(h0, n, rt * 16, k0, T1);
#pragma unroll
    for (int j = 0; j < 6; ++j) {
      v16bf b = fragB_glb(c1t, KCONV, j * 16, k0, 96);
      acc[j] = wmma_bf(a, b, acc[j]);
    }
  }
  int lane = threadIdx.x & 31, half = lane >> 4, cl = lane & 15;
#pragma unroll
  for (int j = 0; j < 6; ++j)
#pragma unroll
    for (int rr = 0; rr < 8; ++rr) {
      int t = rt * 16 + rr + 8 * half;
      int co = j * 16 + cl;
      if (t < T1) out[((long)n * CIN + co) * T1 + t] = acc[j][rr] + bias[co];
    }
}

__global__ void k_conv2(const bhalf* __restrict__ h1, const bhalf* __restrict__ c2t,
                        const float* __restrict__ bias, float* __restrict__ out) {
  int wid = (blockIdx.x * blockDim.x + threadIdx.x) >> 5;
  if (wid >= NSEQ * 2) return;
  int n = wid >> 1, rt = wid & 1;
  v8f acc = {};
  for (int kc = 0; kc < 15; ++kc) {
    int k0 = kc * 32;
    v16bf a = fragA_conv(h1, n, rt * 16, k0, T2);
    v16bf b = fragB_glb(c2t, KCONV, 0, k0, 16);
    acc = wmma_bf(a, b, acc);
  }
  int lane = threadIdx.x & 31, half = lane >> 4, cl = lane & 15;
#pragma unroll
  for (int rr = 0; rr < 8; ++rr) {
    int t = rt * 16 + rr + 8 * half;
    if (t < T2) out[((long)n * DD + cl) * T2 + t] = acc[rr] + bias[cl];
  }
}

// ---------------- K3/K6: per-channel BN statistics ----------------
__global__ void k_stats(const float* __restrict__ src, float* __restrict__ stats,
                        int C, int NT, int Tlen) {
  int c = blockIdx.x;
  float s = 0.f, s2 = 0.f;
  for (int i = threadIdx.x; i < NT; i += blockDim.x) {
    int n = i / Tlen, t = i % Tlen;
    float v = src[((long)n * C + c) * Tlen + t];
    s += v; s2 += v * v;
  }
  __shared__ float rs[256], rq[256];
  rs[threadIdx.x] = s; rq[threadIdx.x] = s2;
  __syncthreads();
  for (int o = 128; o > 0; o >>= 1) {
    if ((int)threadIdx.x < o) { rs[threadIdx.x] += rs[threadIdx.x + o]; rq[threadIdx.x] += rq[threadIdx.x + o]; }
    __syncthreads();
  }
  if (threadIdx.x == 0) { stats[c] = rs[0]; stats[C + c] = rq[0]; }
}

// ---------------- K4: BN1 apply + leaky relu -> bf16 [n][ci][32] ----------------
__global__ void k_bn1(const float* __restrict__ raw, const float* __restrict__ stats,
                      const float* __restrict__ g, const float* __restrict__ be,
                      bhalf* __restrict__ h1) {
  const float Nn = (float)NSEQ * T1;
  int stride = gridDim.x * blockDim.x;
  for (long i = blockIdx.x * blockDim.x + threadIdx.x; i < (long)NSEQ * CIN * T1; i += stride) {
    int t = (int)(i % T1);
    long r = i / T1;
    int c = (int)(r % CIN);
    int n = (int)(r / CIN);
    float mu = stats[c] / Nn;
    float var = stats[CIN + c] / Nn - mu * mu;
    float v = raw[i];
    float hv = (v - mu) * rsqrtf(var + 1e-5f) * g[c] + be[c];
    hv = hv >= 0.f ? hv : 0.01f * hv;
    h1[((long)n * CIN + c) * 32 + t] = f2bf(hv);
  }
}

// ---------------- K7: BN2 apply -> feats f32 [n][t][d] ----------------
__global__ void k_bn2(const float* __restrict__ raw, const float* __restrict__ stats,
                      const float* __restrict__ g, const float* __restrict__ be,
                      float* __restrict__ feats) {
  const float Nn = (float)NSEQ * T2;
  int stride = gridDim.x * blockDim.x;
  for (long i = blockIdx.x * blockDim.x + threadIdx.x; i < (long)NSEQ * DD * T2; i += stride) {
    int t = (int)(i % T2);
    long r = i / T2;
    int c = (int)(r % DD);
    int n = (int)(r / DD);
    float mu = stats[c] / Nn;
    float var = stats[DD + c] / Nn - mu * mu;
    float v = raw[i];
    float hv = (v - mu) * rsqrtf(var + 1e-5f) * g[c] + be[c];
    hv = hv >= 0.f ? hv : 0.01f * hv;
    feats[((long)n * T2 + t) * DD + c] = hv;
  }
}

// ---------------- K8: OAS shrinkage covariance + Cholesky ----------------
__global__ void k_oas(const float* __restrict__ feats, float* __restrict__ xdxo) {
  int n = blockIdx.x;
  int tid = threadIdx.x;
  __shared__ float Xs[T2 * DD];
  __shared__ float mean[DD];
  __shared__ float cov[DD * DD];
  __shared__ float Lm[DD * DD];
  __shared__ float s_shr, s_mu;
  const float* src = feats + (long)n * T2 * DD;
  for (int i = tid; i < T2 * DD; i += 256) Xs[i] = src[i];
  __syncthreads();
  if (tid < DD) {
    float m = 0.f;
    for (int t = 0; t < T2; ++t) m += Xs[t * DD + tid];
    mean[tid] = m / (float)T2;
  }
  __syncthreads();
  {
    int i = tid >> 4, j = tid & 15;
    float e = 0.f;
    for (int t = 0; t < T2; ++t) e += (Xs[t * DD + i] - mean[i]) * (Xs[t * DD + j] - mean[j]);
    cov[tid] = e / (float)(T2 - 1);
  }
  __syncthreads();
  if (tid == 0) {
    float tr = 0.f;
    for (int i = 0; i < DD; ++i) tr += cov[i * 17];
    float mu = tr / (float)DD;
    float alpha = 0.f;
    for (int k = 0; k < DD * DD; ++k) alpha += cov[k] * cov[k];
    alpha /= (float)(DD * DD);
    float num = alpha + mu * mu;
    float den = (float)(T2 + 1) * (alpha - mu * mu / (float)DD);
    s_shr = (den == 0.f) ? 1.0f : fminf(num / den, 1.0f);
    s_mu = mu;
  }
  __syncthreads();
  {
    int i = tid >> 4, j = tid & 15;
    cov[tid] = (1.0f - s_shr) * cov[tid] + ((i == j) ? s_shr * s_mu : 0.0f);
  }
  Lm[tid] = 0.0f;
  __syncthreads();
  for (int j = 0; j < DD; ++j) {
    if (tid == 0) {
      float sum = cov[j * DD + j];
      for (int k = 0; k < j; ++k) sum -= Lm[j * DD + k] * Lm[j * DD + k];
      Lm[j * DD + j] = sqrtf(fmaxf(sum, 1e-20f));
    }
    __syncthreads();
    if (tid > j && tid < DD) {
      float sum = cov[tid * DD + j];
      for (int k = 0; k < j; ++k) sum -= Lm[tid * DD + k] * Lm[j * DD + k];
      Lm[tid * DD + j] = sum / Lm[j * DD + j];
    }
    __syncthreads();
  }
  if (tid < DD) xdxo[(long)n * HID + tid] = Lm[tid * 17];
  if (tid < MOFF) {
    int i = 1, acc = 0;
    while (acc + i <= tid) { acc += i; ++i; }
    int j = tid - acc;
    xdxo[(long)n * HID + DD + tid] = Lm[i * DD + j];
  }
}

// ---------------- K9: persistent scan kernel (1 block, 16 waves) ----------------
// GEMM: M=64 (4 row tiles), up to 16 col tiles, K chunks of 32; fused tanh/Euler epilogue.
template <bool TANH>
static __device__ __forceinline__ void mlp_gemm(const bhalf* A, int lda, int kchunks,
                                                const bhalf* Bw, int ldk, int Nreal, int nmaxB,
                                                const float* bias,
                                                bhalf* obf, int ldo,
                                                float* ohc, float dtf) {
  int wave = threadIdx.x >> 5;
  int rt = wave & 3, cg = wave >> 2;
  int mbase = rt * 16;
  int lane = threadIdx.x & 31;
  int half = lane >> 4, cl = lane & 15;
  int ntiles = (Nreal + 15) >> 4;
  v8f acc[4] = {};
  for (int kc = 0; kc < kchunks; ++kc) {
    int k0 = kc << 5;
    v16bf a = fragA_mem(A, lda, mbase, k0);
#pragma unroll
    for (int j = 0; j < 4; ++j) {
      int ct = cg + 4 * j;
      if (ct < ntiles) {                       // wave-uniform guard: EXEC stays all-ones
        v16bf b = fragB_glb(Bw, ldk, ct * 16, k0, nmaxB);
        acc[j] = wmma_bf(a, b, acc[j]);
      }
    }
  }
#pragma unroll
  for (int j = 0; j < 4; ++j) {
    int ct = cg + 4 * j;
    if (ct >= ntiles) continue;
#pragma unroll
    for (int rr = 0; rr < 8; ++rr) {
      int row = mbase + rr + 8 * half;
      int col = ct * 16 + cl;
      if (col < Nreal) {
        float v = acc[j][rr] + bias[col];
        if (TANH) obf[row * ldo + col] = f2bf(tanhf(v));
        else      ohc[row * 160 + col] += dtf * v;
      }
    }
  }
}

__global__ void k_scan(const float* __restrict__ xdxo,
                       const bhalf* o0t, const bhalf* o1t, const bhalf* o2t, const bhalf* o3t,
                       const float* b0, const float* b1, const float* b2, const float* b3,
                       const float* wxd, const float* whd, const float* bd,
                       const float* wxo, const float* who, const float* bo,
                       const float* clsw, const float* clsb, float* __restrict__ out) {
  extern __shared__ char smem[];
  float* hd = (float*)smem;                   // 64*16
  float* ho = hd + BB * DD;                   // 64*120
  float* hc = ho + BB * MOFF;                 // 64*160 (padded, pad stays 0)
  bhalf* abfA = (bhalf*)(hc + BB * 160);      // 64*256 bf16
  bhalf* abfB = abfA + BB * 256;              // 64*256 bf16
  float* hdn = (float*)abfA;                  // alias: free during GRU phase
  float* hon = (float*)abfB;
  int tid = threadIdx.x;

  for (int i = tid; i < BB * DD; i += 512) hd[i] = 1.0f;
  for (int i = tid; i < BB * MOFF; i += 512) ho[i] = 0.0f;
  for (int i = tid; i < BB * 160; i += 512) hc[i] = 0.0f;
  __syncthreads();

  for (int s = 0; s < SS; ++s) {
    // hc = [log(hd), ho]
    for (int i = tid; i < BB * HID; i += 512) {
      int b = i / HID, q = i % HID;
      hc[b * 160 + q] = (q < DD) ? logf(hd[b * DD + q]) : ho[b * MOFF + (q - DD)];
    }
    __syncthreads();
    // 4 Euler steps of the ODE MLP
    for (int e = 0; e < 4; ++e) {
      for (int i = tid; i < BB * 160; i += 512) abfA[i] = f2bf(hc[i]);
      __syncthreads();
      mlp_gemm<true >(abfA, 160, 5, o0t, 160, ODEU, 256, b0, abfB, 256, nullptr, 0.f);
      __syncthreads();
      mlp_gemm<true >(abfB, 256, 8, o1t, 256, ODEU, 256, b1, abfA, 256, nullptr, 0.f);
      __syncthreads();
      mlp_gemm<true >(abfA, 256, 8, o2t, 256, ODEU, 256, b2, abfB, 256, nullptr, 0.f);
      __syncthreads();
      mlp_gemm<false>(abfB, 256, 8, o3t, 256, HID, 136, b3, (bhalf*)nullptr, 0, hc, 0.25f);
      __syncthreads();
    }
    // unpack
    for (int i = tid; i < BB * HID; i += 512) {
      int b = i / HID, q = i % HID;
      if (q < DD) hd[b * DD + q] = expf(hc[b * 160 + q]);
      else ho[b * MOFF + q - DD] = hc[b * 160 + q];
    }
    __syncthreads();
    // diag (multiplicative) GRU
    for (int i = tid; i < BB * DD; i += 512) {
      int b = i >> 4, ii = i & 15;
      const float* xd = xdxo + ((long)b * SS + s) * HID;
      float gxr = 0, gxi = 0, gxn = 0, ghr = 0, ghi = 0, ghn = 0;
      for (int k = 0; k < DD; ++k) {
        float xv = xd[k], hv = hd[b * DD + k];
        gxr += xv * fabsf(wxd[k * 48 + ii]);
        gxi += xv * fabsf(wxd[k * 48 + 16 + ii]);
        gxn += xv * fabsf(wxd[k * 48 + 32 + ii]);
        ghr += hv * fabsf(whd[k * 48 + ii]);
        ghi += hv * fabsf(whd[k * 48 + 16 + ii]);
        ghn += hv * fabsf(whd[k * 48 + 32 + ii]);
      }
      float br = fabsf(bd[ii]), bi = fabsf(bd[16 + ii]), bn = fabsf(bd[32 + ii]);
      float r = sigm(br * gxr * ghr);
      float z = sigm(bi * gxi * ghi);
      float spx = bn * gxn * r * ghn;
      float ng = (spx > 20.f) ? spx : log1pf(expf(spx));
      float hv = hd[b * DD + ii];
      hdn[i] = expf(z * logf(hv) + (1.f - z) * logf(ng));
    }
    // off-diag GRU
    for (int i = tid; i < BB * MOFF; i += 512) {
      int b = i / MOFF, p = i % MOFF;
      const float* xo = xdxo + ((long)b * SS + s) * HID + DD;
      float gxr = 0, gxi = 0, gxn = 0, ghr = 0, ghi = 0, ghn = 0;
      for (int k = 0; k < MOFF; ++k) {
        float xv = xo[k], hv = ho[b * MOFF + k];
        gxr += xv * wxo[k * 360 + p];
        gxi += xv * wxo[k * 360 + 120 + p];
        gxn += xv * wxo[k * 360 + 240 + p];
        ghr += hv * who[k * 360 + p];
        ghi += hv * who[k * 360 + 120 + p];
        ghn += hv * who[k * 360 + 240 + p];
      }
      float r = sigm(gxr + ghr + bo[p]);
      float z = sigm(gxi + ghi + bo[120 + p]);
      float ng = tanhf(gxn + r * ghn + bo[240 + p]);
      hon[i] = ng + z * (ho[b * MOFF + p] - ng);
    }
    __syncthreads();
    for (int i = tid; i < BB * DD; i += 512) hd[i] = hdn[i];
    for (int i = tid; i < BB * MOFF; i += 512) ho[i] = hon[i];
    __syncthreads();
  }
  // classifier: out = [log hd, ho] @ cls_w + cls_b
  for (int i = tid; i < BB * NCLS; i += 512) {
    int b = i / NCLS, k = i % NCLS;
    float acc = clsb[k];
    for (int q = 0; q < HID; ++q) {
      float f = (q < DD) ? logf(hd[b * DD + q]) : ho[b * MOFF + q - DD];
      acc += f * clsw[q * NCLS + k];
    }
    out[i] = acc;
  }
}

// ---------------- host launcher ----------------
extern "C" void kernel_launch(void* const* d_in, const int* in_sizes, int n_in,
                              void* d_out, int out_size, void* d_ws, size_t ws_size,
                              hipStream_t stream) {
  (void)in_sizes; (void)n_in; (void)out_size; (void)ws_size;
  const float* x    = (const float*)d_in[0];
  const float* fw   = (const float*)d_in[1];
  const float* st   = (const float*)d_in[2];
  const float* c1w  = (const float*)d_in[3];
  const float* c1b  = (const float*)d_in[4];
  const float* bn1g = (const float*)d_in[5];
  const float* bn1b = (const float*)d_in[6];
  const float* c2w  = (const float*)d_in[7];
  const float* c2b  = (const float*)d_in[8];
  const float* bn2g = (const float*)d_in[9];
  const float* bn2b = (const float*)d_in[10];
  const float* wxd  = (const float*)d_in[11];
  const float* whd  = (const float*)d_in[12];
  const float* bd   = (const float*)d_in[13];
  const float* wxo  = (const float*)d_in[14];
  const float* who  = (const float*)d_in[15];
  const float* bo   = (const float*)d_in[16];
  const float* ow0  = (const float*)d_in[17];
  const float* ob0  = (const float*)d_in[18];
  const float* ow1  = (const float*)d_in[19];
  const float* ob1  = (const float*)d_in[20];
  const float* ow2  = (const float*)d_in[21];
  const float* ob2  = (const float*)d_in[22];
  const float* ow3  = (const float*)d_in[23];
  const float* ob3  = (const float*)d_in[24];
  const float* clsw = (const float*)d_in[25];
  const float* clsb = (const float*)d_in[26];

  char* ws = (char*)d_ws;
  bhalf* wfb  = (bhalf*)(ws + OFF_FB);
  bhalf* c1t  = (bhalf*)(ws + OFF_C1T);
  bhalf* c2t  = (bhalf*)(ws + OFF_C2T);
  bhalf* o0t  = (bhalf*)(ws + OFF_O0);
  bhalf* o1t  = (bhalf*)(ws + OFF_O1);
  bhalf* o2t  = (bhalf*)(ws + OFF_O2);
  bhalf* o3t  = (bhalf*)(ws + OFF_O3);
  float* st1  = (float*)(ws + OFF_ST1);
  float* st2  = (float*)(ws + OFF_ST2);
  bhalf* h0   = (bhalf*)(ws + OFF_H0);     // einsum out / conv1 A
  bhalf* h1   = (bhalf*)(ws + OFF_H0);     // BN1 out (h0 dead)
  float* c1r  = (float*)(ws + OFF_BIG);    // conv1 raw
  float* c2r  = (float*)(ws + OFF_C2RAW);  // conv2 raw (c1r dead)
  float* fts  = (float*)(ws + OFF_FEATS);
  float* xdxo = (float*)(ws + OFF_XDXO);

  k_prep  <<<64, 256, 0, stream>>>(fw, st, c1w, c2w, ow0, ow1, ow2, ow3,
                                   wfb, c1t, c2t, o0t, o1t, o2t, o3t);
  k_einsum<<<1200, 256, 0, stream>>>(x, wfb, h0);
  k_conv1 <<<800, 256, 0, stream>>>(h0, c1t, c1b, c1r);
  k_stats <<<96, 256, 0, stream>>>(c1r, st1, CIN, NSEQ * T1, T1);
  k_bn1   <<<2048, 256, 0, stream>>>(c1r, st1, bn1g, bn1b, h1);
  k_conv2 <<<800, 256, 0, stream>>>(h1, c2t, c2b, c2r);
  k_stats <<<16, 256, 0, stream>>>(c2r, st2, DD, NSEQ * T2, T2);
  k_bn2   <<<1024, 256, 0, stream>>>(c2r, st2, bn2g, bn2b, fts);
  k_oas   <<<3200, 256, 0, stream>>>(fts, xdxo);
  k_scan  <<<1, 512, SCAN_LDS_BYTES, stream>>>(xdxo, o0t, o1t, o2t, o3t,
                                               ob0, ob1, ob2, ob3,
                                               wxd, whd, bd, wxo, who, bo,
                                               clsw, clsb, (float*)d_out);
}
